// RBFBasedLatticeUpdateBlockFrac_48404281426064
// MI455X (gfx1250) — matrix operand order repaired and checked by reading, not compiled, not run on hardware
//
#include <hip/hip_runtime.h>
#include <hip/hip_bf16.h>
#include <math.h>

typedef __attribute__((ext_vector_type(16))) __bf16 v16bf;
typedef __attribute__((ext_vector_type(8)))  __bf16 v8bf;
typedef __attribute__((ext_vector_type(8)))  float  v8f;

#define NGRAPH 256
#define EMB    64
#define EMBR   16

// ---------------- zero output + counts (must run every call) ----------------
__global__ void zero_out_kernel(float* __restrict__ out, float* __restrict__ counts) {
  for (int k = threadIdx.x; k < NGRAPH * 9; k += blockDim.x) out[k] = 0.0f;
  for (int k = threadIdx.x; k < NGRAPH;     k += blockDim.x) counts[k] = 0.0f;
}

// ---------------- per-graph edge counts (LDS histogram) ----------------
__global__ void edge_count_kernel(const long long* __restrict__ edge_src,
                                  const long long* __restrict__ batch,
                                  float* __restrict__ counts, int nEdges) {
  __shared__ float hist[NGRAPH];
  for (int i = threadIdx.x; i < NGRAPH; i += blockDim.x) hist[i] = 0.0f;
  __syncthreads();
  const int stride = blockDim.x * gridDim.x;
  for (int e = blockIdx.x * blockDim.x + threadIdx.x; e < nEdges; e += stride) {
    int src = (int)edge_src[e];
    int g   = (int)batch[src];
    atomicAdd(&hist[g], 1.0f);
  }
  __syncthreads();
  for (int i = threadIdx.x; i < NGRAPH; i += blockDim.x) {
    float v = hist[i];
    if (v != 0.0f) atomicAdd(&counts[i], v);
  }
}

// ---------------- fused edge MLP (bf16 WMMA) + outer-product segment sum ----------------
__launch_bounds__(256)
__global__ void fused_edge_kernel(const float* __restrict__ edge_emb,
                                  const long long* __restrict__ edge_src,
                                  const float* __restrict__ dvec,
                                  const long long* __restrict__ batch,
                                  const float* __restrict__ rbf,
                                  const float* __restrict__ W1,
                                  const float* __restrict__ W2,
                                  const float* __restrict__ Wrbf,
                                  const float* __restrict__ Wout,
                                  const float* __restrict__ counts,
                                  float* __restrict__ out,
                                  int nEdges) {
  __shared__ __attribute__((aligned(32))) __bf16 w1s[EMB * EMB];   // [n][k]
  __shared__ __attribute__((aligned(32))) __bf16 w2s[EMB * EMB];   // [n][k]
  __shared__ __attribute__((aligned(32))) __bf16 wrs[EMB * 32];    // [n][k padded K=32]
  __shared__ float wouts[EMB];
  __shared__ __attribute__((aligned(32))) __bf16 hstage[8][16 * EMB]; // per-wave h tile [m][k]
  __shared__ float scores[8][16];
  __shared__ float acc[NGRAPH * 9];

  const int tid = threadIdx.x;

  // Stage weights to LDS as bf16, B-layout-friendly (row n, contiguous k).
  for (int i = tid; i < EMB * EMB; i += blockDim.x) {
    int n = i >> 6, k = i & 63;
    w1s[i] = (__bf16)W1[k * EMB + n];
    w2s[i] = (__bf16)W2[k * EMB + n];
  }
  for (int i = tid; i < EMB * 32; i += blockDim.x) {
    int n = i >> 5, k = i & 31;
    wrs[i] = (k < EMBR) ? (__bf16)Wrbf[k * EMB + n] : (__bf16)0.0f;
  }
  for (int i = tid; i < EMB; i += blockDim.x) wouts[i] = Wout[i];
  for (int i = tid; i < NGRAPH * 9; i += blockDim.x) acc[i] = 0.0f;
  __syncthreads();

  const int lane  = tid & 31;
  const int wv    = tid >> 5;
  const int half  = lane >> 4;   // 0: lanes 0-15, 1: lanes 16-31
  const int lm    = lane & 15;
  const int abase = half * 8;    // A-fragment K sub-base (16-bit A layout)
  const int bbase = half * 16;   // unused; B fragments also use abase

  // Build resident B fragments (loop-invariant).
  v16bf b1[2][4], b2[2][4], br[4];
#pragma unroll
  for (int t = 0; t < 2; ++t)
#pragma unroll
    for (int j = 0; j < 4; ++j) {
      const int n = j * 16 + lm;
      b1[t][j] = __builtin_shufflevector(*(const v8bf*)&w1s[n * EMB + t * 32 + abase], *(const v8bf*)&w1s[n * EMB + t * 32 + 16 + abase], 0, 1, 2, 3, 4, 5, 6, 7, 8, 9, 10, 11, 12, 13, 14, 15);
      b2[t][j] = __builtin_shufflevector(*(const v8bf*)&w2s[n * EMB + t * 32 + abase], *(const v8bf*)&w2s[n * EMB + t * 32 + 16 + abase], 0, 1, 2, 3, 4, 5, 6, 7, 8, 9, 10, 11, 12, 13, 14, 15);
    }
#pragma unroll
  for (int j = 0; j < 4; ++j)
    br[j] = __builtin_shufflevector(*(const v8bf*)&wrs[(j * 16 + lm) * 32 + abase], *(const v8bf*)&wrs[(j * 16 + lm) * 32 + 16 + abase], 0, 1, 2, 3, 4, 5, 6, 7, 8, 9, 10, 11, 12, 13, 14, 15);

  const v8f vzero = {};
  const int nTiles = (nEdges + 127) >> 7;  // 128 edges per block-iteration (8 waves x 16)

  for (int tile = blockIdx.x; tile < nTiles; tile += (int)gridDim.x) {
    const int e0 = tile * 128 + wv * 16;
    int eRow = e0 + lm;
    if (eRow > nEdges - 1) eRow = nEdges - 1;
    const float* arow = edge_emb + (size_t)eRow * EMB;

    // Speculative L2 prefetch of this wave's rows for the next tile.
    {
      long nxt = (long)(tile + (int)gridDim.x) * 128 + wv * 16 + lm;
      if (nxt < nEdges) __builtin_prefetch(edge_emb + (size_t)nxt * EMB, 0, 1);
    }

    v8f c[4];
#pragma unroll
    for (int j = 0; j < 4; ++j) c[j] = vzero;

    // ---- GEMM1: h = X @ W1 (K=64 in two 16x16x32 steps) ----
#pragma unroll
    for (int t = 0; t < 2; ++t) {
      const float4 f0 = *(const float4*)(arow + t * 32 + abase);
      const float4 f1 = *(const float4*)(arow + t * 32 + abase + 4);
      const float4 g0 = *(const float4*)(arow + t * 32 + 16 + abase);
      const float4 g1 = *(const float4*)(arow + t * 32 + 16 + abase + 4);
      v16bf a;
      a[0]=(__bf16)f0.x; a[1]=(__bf16)f0.y; a[2] =(__bf16)f0.z; a[3] =(__bf16)f0.w;
      a[4]=(__bf16)f1.x; a[5]=(__bf16)f1.y; a[6] =(__bf16)f1.z; a[7] =(__bf16)f1.w;
      a[8]=(__bf16)g0.x; a[9]=(__bf16)g0.y; a[10]=(__bf16)g0.z; a[11]=(__bf16)g0.w;
      a[12]=(__bf16)g1.x; a[13]=(__bf16)g1.y; a[14]=(__bf16)g1.z; a[15]=(__bf16)g1.w;
#pragma unroll
      for (int j = 0; j < 4; ++j)
        c[j] = __builtin_amdgcn_wmma_f32_16x16x32_bf16(false, a, false, b1[t][j],
                                                       (short)0, c[j], false, false);
    }

    // ---- SiLU (fast: exp + v_rcp_f32), restage h as bf16 in per-wave LDS ----
#pragma unroll
    for (int j = 0; j < 4; ++j) {
      const int n = j * 16 + lm;
#pragma unroll
      for (int r = 0; r < 8; ++r) {
        const float x   = c[j][r];
        const float sig = __builtin_amdgcn_rcpf(1.0f + __expf(-x));
        hstage[wv][(r + half * 8) * EMB + n] = (__bf16)(x * sig);
      }
    }
    asm volatile("s_wait_dscnt 0" ::: "memory");  // same-wave LDS store->load

    // ---- GEMM2: x_F = h @ W2 ----
#pragma unroll
    for (int j = 0; j < 4; ++j) c[j] = vzero;
#pragma unroll
    for (int t = 0; t < 2; ++t) {
      const v8bf h0 = *(const v8bf*)&hstage[wv][lm * EMB + t * 32 + abase];
      const v8bf h1 = *(const v8bf*)&hstage[wv][lm * EMB + t * 32 + 16 + abase];
      v16bf a;
#pragma unroll
      for (int i = 0; i < 8; ++i) { a[i] = h0[i]; a[8 + i] = h1[i]; }
#pragma unroll
      for (int j = 0; j < 4; ++j)
        c[j] = __builtin_amdgcn_wmma_f32_16x16x32_bf16(false, a, false, b2[t][j],
                                                       (short)0, c[j], false, false);
    }

    // ---- rbf GEMM: g = rbf @ W_rbf (K=16 zero-padded to 32) ----
    const float* rrow = rbf + (size_t)eRow * EMBR;
    const float4 r0 = *(const float4*)(rrow + abase);
    const float4 r1 = *(const float4*)(rrow + abase + 4);
    v16bf ar;
    ar[0]=(__bf16)r0.x; ar[1]=(__bf16)r0.y; ar[2]=(__bf16)r0.z; ar[3]=(__bf16)r0.w;
    ar[4]=(__bf16)r1.x; ar[5]=(__bf16)r1.y; ar[6]=(__bf16)r1.z; ar[7]=(__bf16)r1.w;
#pragma unroll
    for (int i = 8; i < 16; ++i) ar[i] = (__bf16)0.0f;
    v8f g4[4];
#pragma unroll
    for (int j = 0; j < 4; ++j) {
      g4[j] = vzero;
      g4[j] = __builtin_amdgcn_wmma_f32_16x16x32_bf16(false, ar, false, br[j],
                                                      (short)0, g4[j], false, false);
    }

    // ---- gate + dot with W_out, cross-lane reduce over N ----
    float part[8];
#pragma unroll
    for (int r = 0; r < 8; ++r) part[r] = 0.0f;
#pragma unroll
    for (int j = 0; j < 4; ++j) {
      const float w = wouts[j * 16 + lm];
#pragma unroll
      for (int r = 0; r < 8; ++r) part[r] += (c[j][r] * g4[j][r]) * w;
    }
#pragma unroll
    for (int m = 1; m < 16; m <<= 1)
#pragma unroll
      for (int r = 0; r < 8; ++r) part[r] += __shfl_xor(part[r], m, 32);
    if (lm == 0) {
#pragma unroll
      for (int r = 0; r < 8; ++r) scores[wv][half * 8 + r] = part[r];
    }
    asm volatile("s_wait_dscnt 0" ::: "memory");

    // ---- per-edge finalize: normalize, symmetric outer product, LDS segment-sum ----
    if (lane < 16) {
      const int e = e0 + lane;
      if (e < nEdges) {
        const int src = (int)edge_src[e];
        const int g   = (int)batch[src];
        const float s = scores[wv][lane] * __builtin_amdgcn_rcpf(counts[g]);
        const float dx = dvec[3 * e], dy = dvec[3 * e + 1], dz = dvec[3 * e + 2];
        const float inv = __builtin_amdgcn_rsqf(dx * dx + dy * dy + dz * dz);
        const float d[3] = {dx, dy, dz};
        const float u[3] = {dx * inv, dy * inv, dz * inv};
        const float hs = 0.5f * s;
        float* ag = &acc[g * 9];
#pragma unroll
        for (int i = 0; i < 3; ++i)
#pragma unroll
          for (int j = 0; j < 3; ++j)
            atomicAdd(&ag[i * 3 + j], hs * (u[i] * d[j] + u[j] * d[i]));
      }
    }
  }

  // ---- flush block-local accumulator ----
  __syncthreads();
  for (int i = tid; i < NGRAPH * 9; i += blockDim.x)
    atomicAdd(&out[i], acc[i]);
}

extern "C" void kernel_launch(void* const* d_in, const int* in_sizes, int n_in,
                              void* d_out, int out_size, void* d_ws, size_t ws_size,
                              hipStream_t stream) {
  (void)n_in; (void)out_size; (void)ws_size;
  const float*     edge_emb   = (const float*)d_in[0];
  const long long* edge_index = (const long long*)d_in[1];  // row 0 = src nodes
  const float*     dvec       = (const float*)d_in[2];
  // d_in[3] = lattice (unused by reference)
  const long long* batch      = (const long long*)d_in[4];
  const float*     rbf        = (const float*)d_in[5];
  const float*     W1         = (const float*)d_in[6];
  const float*     W2         = (const float*)d_in[7];
  const float*     Wrbf       = (const float*)d_in[8];
  const float*     Wout       = (const float*)d_in[9];
  float* out    = (float*)d_out;
  float* counts = (float*)d_ws;  // 256 floats of scratch
  const int nEdges = in_sizes[0] / EMB;

  zero_out_kernel<<<1, 256, 0, stream>>>(out, counts);
  edge_count_kernel<<<512, 256, 0, stream>>>(edge_index, batch, counts, nEdges);
  fused_edge_kernel<<<512, 256, 0, stream>>>(edge_emb, edge_index, dvec, batch, rbf,
                                             W1, W2, Wrbf, Wout, counts, out, nEdges);
}
